// VQLayer_88751204205147
// MI455X (gfx1250) — matrix operand order repaired and checked by reading, not compiled, not run on hardware
//
#include <hip/hip_runtime.h>
#include <math.h>

// ---------------------------------------------------------------------------
// VQ-VAE vector quantization for MI455X (gfx1250, wave32, WMMA).
//   x:          [32,32,32,512] f32  -> N=32768 rows of D=512
//   embeddings: [512, 4096]    f32
// Outputs (concat in d_out): quantized_st [N*D] f32, loss f32, perplexity f32
//
// GEMM path: split-bf16 (hi/lo) WMMA 16x16x32, 3 products per chunk for
// ~fp32-accurate dot products; argmin fused, distances never materialized.
// E is pre-converted/pre-swizzled once into per-lane fragment order, so the
// hot loop stages B tiles with async global->LDS copies (double-buffered).
// ---------------------------------------------------------------------------

#define DD      512
#define KK      4096
#define NROWS   32768
#define NUMEL   (NROWS * DD)      // 16777216
#define CHUNKS  (DD / 32)         // 16 D-chunks of 32
#define KTILE   16                // codes per WMMA tile
#define NT      (KK / KTILE)      // 256 K-tiles
#define TILE_US (DD * KTILE)      // 8192 ushorts = 16KB per (hi|lo) tile
#define RPB     128               // rows per block (8 waves * 16 rows)

typedef __attribute__((ext_vector_type(16))) __bf16 v16bf;
typedef __attribute__((ext_vector_type(8)))  float  v8f;

// 16-byte payload type matching the async-to-LDS builtin's signature
// (hipcc reported: 'int __attribute__((vector_size(16))) __device__ *').
typedef int b128_t __attribute__((vector_size(16)));
typedef __attribute__((address_space(1))) b128_t* gptr_b128;
typedef __attribute__((address_space(3))) b128_t* lptr_b128;

#if defined(__has_builtin)
#if __has_builtin(__builtin_amdgcn_global_load_async_to_lds_b128) && \
    __has_builtin(__builtin_amdgcn_s_wait_asynccnt)
#define USE_ASYNC_LDS 1
#endif
#endif

__device__ __forceinline__ unsigned short f32_to_bf16_rne(float f) {
  unsigned u = __builtin_bit_cast(unsigned, f);
  unsigned r = u + 0x7FFFu + ((u >> 16) & 1u);
  return (unsigned short)(r >> 16);
}
__device__ __forceinline__ float bf16_bits_to_f32(unsigned short h) {
  return __builtin_bit_cast(float, ((unsigned)h) << 16);
}
__device__ __forceinline__ __bf16 bf16_from_bits(unsigned short h) {
  return __builtin_bit_cast(__bf16, h);
}

// ---------------------------------------------------------------------------
// counts[k] = 0
__global__ void vq_init_kernel(unsigned* __restrict__ counts) {
  int t = blockIdx.x * blockDim.x + threadIdx.x;
  if (t < KK) counts[t] = 0u;
}

// colnorm[k] = sum_d E[d,k]^2   (coalesced over k)
__global__ void vq_colnorm_kernel(const float* __restrict__ E,
                                  float* __restrict__ cn) {
  int k = blockIdx.x * blockDim.x + threadIdx.x;
  if (k >= KK) return;
  float s = 0.f;
  for (int d = 0; d < DD; ++d) {
    float v = E[(size_t)d * KK + k];
    s = fmaf(v, v, s);
  }
  cn[k] = s;
}

// One-time E conversion: f32 -> bf16 hi/lo, written in per-lane B-fragment
// order, grouped per K-tile (16KB contiguous per tile per array).
// slot = chunk*32 + (dlo&16) + n holds 16 consecutive-d bf16 (32B per lane).
__global__ __launch_bounds__(256)
void vq_prep_e_kernel(const float* __restrict__ E,
                      unsigned short* __restrict__ e_hi,
                      unsigned short* __restrict__ e_lo) {
  int idx = blockIdx.x * 256 + threadIdx.x;   // idx = d*KK + k
  float f = E[idx];
  unsigned short h  = f32_to_bf16_rne(f);
  unsigned short lo = f32_to_bf16_rne(f - bf16_bits_to_f32(h));
  int d   = idx >> 12;
  int k   = idx & (KK - 1);
  int kt  = k >> 4;
  int n   = k & 15;
  int dlo = d & 31;
  int pos = kt * TILE_US + (((d >> 5) * 32) + (dlo & 16) + n) * 16 + (dlo & 15);
  e_hi[pos] = h;
  e_lo[pos] = lo;
}

// ---------------------------------------------------------------------------
// Stage one 16KB bf16 tile (global, pre-swizzled) into LDS.
// 256 threads x 4 x b128; consecutive lanes touch consecutive 16B segments.
__device__ __forceinline__ void stage_tile(const unsigned short* __restrict__ g,
                                           unsigned short* l, int tid) {
#ifdef USE_ASYNC_LDS
#pragma unroll
  for (int s = 0; s < 4; ++s) {
    int off = (s * 256 + tid) * 8;     // 8 ushorts = 16B
    __builtin_amdgcn_global_load_async_to_lds_b128(
        (gptr_b128)(g + off), (lptr_b128)(l + off), 0, 0);
  }
#else
#pragma unroll
  for (int s = 0; s < 4; ++s) {
    int off = (s * 256 + tid) * 8;
    *reinterpret_cast<uint4*>(l + off) =
        *reinterpret_cast<const uint4*>(g + off);
  }
#endif
}

// ---------------------------------------------------------------------------
// Main fused GEMM + argmin. One wave owns 16 rows; A (x rows) resident in
// VGPRs as hi/lo bf16 fragments; B (E tile) double-buffered in LDS via
// async global->LDS copies.
__global__ __launch_bounds__(256, 1)
void vq_argmin_kernel(const float* __restrict__ x,
                      const unsigned short* __restrict__ e_hi,
                      const unsigned short* __restrict__ e_lo,
                      const float* __restrict__ colnorm,
                      int* __restrict__ outIdx) {
  __shared__ __align__(32) unsigned short lds_hi[2][TILE_US];
  __shared__ __align__(32) unsigned short lds_lo[2][TILE_US];

  const int tid  = threadIdx.x;
  const int wave = tid >> 5;
  const int lane = tid & 31;
  const int half = lane >> 4;      // 0: lanes 0-15, 1: lanes 16-31
  const int l    = lane & 15;
  const int rowBase = blockIdx.x * RPB + wave * 16;
  const int row     = rowBase + l;

  // --- Kick off tile 0 staging before the (long) A-fragment load ---------
  stage_tile(e_hi, &lds_hi[0][0], tid);
  stage_tile(e_lo, &lds_lo[0][0], tid);

  // --- Load this wave's 16 rows as resident A fragments (hi/lo split) ----
  // A k order used by this code (same as packed E): lanes 0-15 hold K={0..15}, lanes 16-31
  // hold K={16..31}; v16bf element e -> the e-th of those K values.
  v16bf a_hi[CHUNKS], a_lo[CHUNKS];
#pragma unroll
  for (int c = 0; c < CHUNKS; ++c) {
    const float* p0 = x + (size_t)row * DD + c * 32 + half * 16; // K group 1
    const float* p1 = p0 + 8;                                    // K group 2
#pragma unroll
    for (int j = 0; j < 8; ++j) {
      float f0 = p0[j], f1 = p1[j];
      unsigned short h0 = f32_to_bf16_rne(f0);
      unsigned short h1 = f32_to_bf16_rne(f1);
      a_hi[c][j]     = bf16_from_bits(h0);
      a_hi[c][8 + j] = bf16_from_bits(h1);
      a_lo[c][j]     = bf16_from_bits(f32_to_bf16_rne(f0 - bf16_bits_to_f32(h0)));
      a_lo[c][8 + j] = bf16_from_bits(f32_to_bf16_rne(f1 - bf16_bits_to_f32(h1)));
    }
  }

  float bestd[8];
  int   besti[8];
#pragma unroll
  for (int i = 0; i < 8; ++i) { bestd[i] = 3.4e38f; besti[i] = 0; }

  for (int kt = 0; kt < NT; ++kt) {
    const int kbase = kt * KTILE;
    const int buf   = kt & 1;

    // Prefetch next tile into the other LDS buffer (overlaps with compute).
    if (kt + 1 < NT) {
      stage_tile(e_hi + (size_t)(kt + 1) * TILE_US, &lds_hi[buf ^ 1][0], tid);
      stage_tile(e_lo + (size_t)(kt + 1) * TILE_US, &lds_lo[buf ^ 1][0], tid);
#ifdef USE_ASYNC_LDS
      __builtin_amdgcn_s_wait_asynccnt(8);   // tile kt's 8 copies complete
#endif
    } else {
#ifdef USE_ASYNC_LDS
      __builtin_amdgcn_s_wait_asynccnt(0);
#endif
    }
    __syncthreads();   // all waves' portions of tile kt are in LDS

    // --- 16x16 dot-product tile: 48 WMMAs (hi*hi + hi*lo + lo*hi) --------
    v8f acc = {};
#pragma unroll
    for (int c = 0; c < CHUNKS; ++c) {
      const v16bf b_hi = *reinterpret_cast<const v16bf*>(&lds_hi[buf][(c * 32 + lane) * 16]);
      const v16bf b_lo = *reinterpret_cast<const v16bf*>(&lds_lo[buf][(c * 32 + lane) * 16]);
      acc = __builtin_amdgcn_wmma_f32_16x16x32_bf16(false, a_hi[c], false, b_hi,
                                                    (short)0, acc, false, false);
      acc = __builtin_amdgcn_wmma_f32_16x16x32_bf16(false, a_hi[c], false, b_lo,
                                                    (short)0, acc, false, false);
      acc = __builtin_amdgcn_wmma_f32_16x16x32_bf16(false, a_lo[c], false, b_hi,
                                                    (short)0, acc, false, false);
    }

    // C/D layout: lane -> col N = lane%16; VGPR i -> row M = i + 8*half.
    const float cn    = colnorm[kbase + l];
    const int   myIdx = kbase + l;
#pragma unroll
    for (int i = 0; i < 8; ++i) {
      float dist = cn - 2.0f * acc[i];   // |e|^2 - 2 x.e  (|x|^2 dropped)
      if (dist < bestd[i]) { bestd[i] = dist; besti[i] = myIdx; }
    }
    __syncthreads();   // done reading buf before it is overwritten
  }

  // --- Argmin across the 16 lanes holding each row (first-index ties) ----
#pragma unroll
  for (int off = 8; off >= 1; off >>= 1) {
#pragma unroll
    for (int i = 0; i < 8; ++i) {
      float od = __shfl_xor(bestd[i], off, 16);
      int   oi = __shfl_xor(besti[i], off, 16);
      if (od < bestd[i] || (od == bestd[i] && oi < besti[i])) {
        bestd[i] = od; besti[i] = oi;
      }
    }
  }
  if (l == 0) {
#pragma unroll
    for (int i = 0; i < 8; ++i)
      outIdx[rowBase + half * 8 + i] = besti[i];
  }
}

// ---------------------------------------------------------------------------
// Gather winning code, emit quantized_st = x + (q - x) exactly as reference,
// per-block partial sum of (q-x)^2 (deterministic), histogram counts.
__global__ __launch_bounds__(256)
void vq_gather_kernel(const float* __restrict__ x,
                      const float* __restrict__ E,
                      const int* __restrict__ idxArr,
                      float* __restrict__ outQ,
                      float* __restrict__ partial,
                      unsigned* __restrict__ counts) {
  __shared__ float red[256];
  const int row  = blockIdx.x;
  const int t    = threadIdx.x;
  const int code = idxArr[row];
  if (t == 0) atomicAdd(&counts[code], 1u);

  const float* xr = x + (size_t)row * DD;
  float*       qr = outQ + (size_t)row * DD;
  float s = 0.f;
#pragma unroll
  for (int j = 0; j < 2; ++j) {
    int d = t + j * 256;
    float q    = E[(size_t)d * KK + code];   // E column (L2-resident)
    float xv   = xr[d];
    float diff = q - xv;
    qr[d] = xv + diff;                       // straight-through forward value
    s = fmaf(diff, diff, s);
  }
  red[t] = s;
  __syncthreads();
  for (int off = 128; off >= 1; off >>= 1) {
    if (t < off) red[t] += red[t + off];
    __syncthreads();
  }
  if (t == 0) partial[row] = red[0];
}

// ---------------------------------------------------------------------------
// loss = 1.25 * mean((q-x)^2); perplexity = exp(-sum p*log(p+1e-10))
__global__ __launch_bounds__(256)
void vq_finalize_kernel(const unsigned* __restrict__ counts,
                        const float* __restrict__ partial,
                        float* __restrict__ outScalars) {
  __shared__ float redH[256];
  __shared__ float redS[256];
  const int t = threadIdx.x;

  float H = 0.f;
  for (int k = t; k < KK; k += 256) {
    float p = (float)counts[k] * (1.0f / (float)NROWS);
    H += p * logf(p + 1e-10f);
  }
  float S = 0.f;
  for (int r = t; r < NROWS; r += 256) S += partial[r];

  redH[t] = H; redS[t] = S;
  __syncthreads();
  for (int off = 128; off >= 1; off >>= 1) {
    if (t < off) { redH[t] += redH[t + off]; redS[t] += redS[t + off]; }
    __syncthreads();
  }
  if (t == 0) {
    float m = redS[0] / (float)NUMEL;
    outScalars[0] = m + 0.25f * m;   // q_latent + commitment * e_latent
    outScalars[1] = expf(-redH[0]);
  }
}

// ---------------------------------------------------------------------------
extern "C" void kernel_launch(void* const* d_in, const int* in_sizes, int n_in,
                              void* d_out, int out_size, void* d_ws, size_t ws_size,
                              hipStream_t stream) {
  const float* x = (const float*)d_in[0];   // [NROWS, DD]
  const float* E = (const float*)d_in[1];   // [DD, KK]
  float* out = (float*)d_out;               // [NUMEL] quantized_st, +loss, +perp

  char* ws = (char*)d_ws;
  unsigned short* e_hi    = (unsigned short*)(ws);             // 4 MB
  unsigned short* e_lo    = (unsigned short*)(ws + 4194304);   // 4 MB
  float*          colnorm = (float*)(ws + 8388608);            // 16 KB
  int*            indices = (int*)(ws + 8404992);              // 128 KB
  unsigned*       counts  = (unsigned*)(ws + 8536064);         // 16 KB
  float*          partial = (float*)(ws + 8552448);            // 128 KB

  vq_init_kernel<<<(KK + 255) / 256, 256, 0, stream>>>(counts);
  vq_colnorm_kernel<<<(KK + 255) / 256, 256, 0, stream>>>(E, colnorm);
  vq_prep_e_kernel<<<(DD * KK) / 256, 256, 0, stream>>>(E, e_hi, e_lo);
  vq_argmin_kernel<<<NROWS / RPB, 256, 0, stream>>>(x, e_hi, e_lo, colnorm, indices);
  vq_gather_kernel<<<NROWS, 256, 0, stream>>>(x, E, indices, out, partial, counts);
  vq_finalize_kernel<<<1, 256, 0, stream>>>(counts, partial, out + NUMEL);
}
